// Model_7928509628900
// MI455X (gfx1250) — matrix operand order repaired and checked by reading, not compiled, not run on hardware
//
#include <hip/hip_runtime.h>
#include <hip/hip_bf16.h>

// ---------------------------------------------------------------------------
// PredNet forward for MI455X (gfx1250).
// All 3x3 convs are implicit GEMMs on v_wmma_f32_16x16x32_f16 (wave32).
// K' ordering is tap-major, each source's channels padded to 16, and groups-
// per-tap padded to even so each tap owns exactly cpt K-chunks: the tap loop
// is outer (9 iterations, all tap/bounds/pixel math hoisted, zero divisions
// in the kernel), and the inner per-chunk work is just a 2-compare source
// select + 16 fixed-stride f16 gathers + aligned 32B A-fragment vector loads
// + WMMAs. Weights are pre-packed once per launch into the identical K'
// permutation in CDNA5 A-fragment order.
// ---------------------------------------------------------------------------

typedef __attribute__((ext_vector_type(16))) _Float16 v16h;
typedef __attribute__((ext_vector_type(8)))  float    v8f;

__device__ __forceinline__ float hsig(float x) {
    return fminf(fmaxf(0.2f * x + 0.5f, 0.f), 1.f);
}

// A fragment: pre-packed [tile][chunk][lane][16] f16 -> one 32B vector load.
__device__ __forceinline__ v16h load_packed_a(const _Float16* __restrict__ packed,
                                              int tile, int nChunks, int ck, int lane) {
    const _Float16* p = packed + (((size_t)tile * nChunks + ck) * 32 + lane) * 16;
    return *reinterpret_cast<const v16h*>(p);
}

// ---------------------------------------------------------------------------
// Weight repack: OIHW f32 -> [tile][chunk][lane][16] f16, A-fragment order,
// in the SAME tap-major / 16-padded / even-padded K' permutation.
// tile = gate*MT + mt; row m is oc = gate*R + mt*16 + m.
// Lanes 0-15 (row m) hold K={0..15}; lanes 16-31 hold {16..31}.
// ---------------------------------------------------------------------------
__global__ void pack_w_kernel(const float* __restrict__ Wt, _Float16* __restrict__ outp,
                              int G, int R, int MT,
                              int n0, int n1, int n2, int gp0, int gp1, int gpt,
                              int gptPad, int nChunks, int total)
{
    const int idx = blockIdx.x * blockDim.x + threadIdx.x;
    if (idx >= total) return;
    const int e    = idx & 15;
    const int lane = (idx >> 4) & 31;
    const int ck   = (idx >> 9) % nChunks;
    const int tile = (idx >> 9) / nChunks;
    const int gate = tile / MT;
    const int mt   = tile - gate * MT;
    const int m    = lane & 15;
    const bool hi  = lane >= 16;
    const int kin  = (hi ? 16 : 0) + e;
    const int kp   = ck * 32 + kin;          // K' index
    const int grp  = kp >> 4, off = kp & 15;
    const int tap  = grp / gptPad;           // one-time kernel: divisions OK here
    const int sg   = grp - tap * gptPad;
    const int Cin  = n0 + n1 + n2;
    const int ch   = mt * 16 + m;

    float v = 0.f;
    if (tap < 9 && sg < gpt && ch < R && gate < G) {
        int ic = -1;
        if (sg < gp0)            { const int icl = sg * 16 + off;               if (icl < n0) ic = icl; }
        else if (sg < gp0 + gp1) { const int icl = (sg - gp0) * 16 + off;       if (icl < n1) ic = n0 + icl; }
        else                     { const int icl = (sg - gp0 - gp1) * 16 + off; if (icl < n2) ic = n0 + n1 + icl; }
        if (ic >= 0) v = Wt[((size_t)(gate * R + ch) * Cin + ic) * 9 + tap];
    }
    outp[idx] = (_Float16)v;
}

// ---------------------------------------------------------------------------
// Fused ConvLSTM: gates = conv3x3([Rep, Err, up2(RepTop)]) + b; one wave does
// 16 rep-channels x 16 pixels; 4 WMMAs per K-chunk share one B fragment.
// ---------------------------------------------------------------------------
__global__ __launch_bounds__(32)
void convlstm_wmma(const _Float16* __restrict__ rep,
                   const _Float16* __restrict__ err,
                   const _Float16* __restrict__ reptop,
                   const _Float16* __restrict__ packedW,   // tiles: [4*MT][nCh][32][16]
                   const float* __restrict__ bias,         // [4R]
                   float* __restrict__ cell,               // [B,R,H,W] f32 in/out
                   _Float16* __restrict__ repOut,          // [B,R,H,W] f16
                   int R, int S2, int Rt,
                   int gp0, int gp1, int gpt, int cpt,
                   int H, int W, int MT)
{
    const int lane  = threadIdx.x;
    const int HW    = H * W;
    const int ptile = blockIdx.x;
    const int mtile = blockIdx.y;
    const int nChunks = 9 * cpt;

    const int n       = lane & 15;
    const int halfIdx = (lane >= 16) ? 1 : 0;
    const int p0 = ptile * 16;               // 16 flat pixels; never cross batch
    const int b  = p0 / HW;
    const int hw = (p0 - b * HW) + n;        // this lane's pixel (fixed)
    const int h  = hw / W, w = hw - h * W;

    const int H2 = H >> 1, W2 = W >> 1;
    const size_t strF = (size_t)HW;
    const size_t strH = (size_t)H2 * W2;
    const _Float16* base0 = rep + (size_t)b * R  * strF;
    const _Float16* base1 = err + (size_t)b * S2 * strF;
    const _Float16* base2 = reptop ? reptop + (size_t)b * Rt * strH : nullptr;

    v8f acc0 = {}, acc1 = {}, acc2 = {}, acc3 = {};
    for (int tap = 0; tap < 9; ++tap) {
        const int ky = (tap >= 6) ? 2 : (tap >= 3) ? 1 : 0;
        const int kx = tap - ky * 3;
        const int hh = h + ky - 1, ww = w + kx - 1;
        const bool inb = ((unsigned)hh < (unsigned)H) && ((unsigned)ww < (unsigned)W);
        const size_t pixF = (size_t)hh * W + ww;
        const size_t pixH = (size_t)(hh >> 1) * W2 + (ww >> 1);

#pragma unroll 2
        for (int sp = 0; sp < cpt; ++sp) {
            const int ck = tap * cpt + sp;
            const int sg = 2 * sp + halfIdx;
            v16h bf = {};
            if (inb && sg < gpt) {
                const _Float16* sptr;
                size_t cstride;
                int nvalid;
                if (sg < gp0) {
                    sptr = base0 + (size_t)(sg * 16) * strF + pixF;
                    cstride = strF;  nvalid = R - sg * 16;
                } else if (sg < gp0 + gp1) {
                    const int icb = (sg - gp0) * 16;
                    sptr = base1 + (size_t)icb * strF + pixF;
                    cstride = strF;  nvalid = S2 - icb;
                } else {
                    const int icb = (sg - gp0 - gp1) * 16;
                    sptr = base2 + (size_t)icb * strH + pixH;
                    cstride = strH;  nvalid = Rt - icb;
                }
                if (nvalid > 16) nvalid = 16;
#pragma unroll
                for (int e = 0; e < 16; ++e)
                    if (e < nvalid) bf[e] = sptr[(size_t)e * cstride];
            }
            const v16h a0 = load_packed_a(packedW, 0 * MT + mtile, nChunks, ck, lane);
            const v16h a1 = load_packed_a(packedW, 1 * MT + mtile, nChunks, ck, lane);
            const v16h a2 = load_packed_a(packedW, 2 * MT + mtile, nChunks, ck, lane);
            const v16h a3 = load_packed_a(packedW, 3 * MT + mtile, nChunks, ck, lane);
            acc0 = __builtin_amdgcn_wmma_f32_16x16x32_f16(false, a0, false, bf, (short)0, acc0, false, false);
            acc1 = __builtin_amdgcn_wmma_f32_16x16x32_f16(false, a1, false, bf, (short)0, acc1, false, false);
            acc2 = __builtin_amdgcn_wmma_f32_16x16x32_f16(false, a2, false, bf, (short)0, acc2, false, false);
            acc3 = __builtin_amdgcn_wmma_f32_16x16x32_f16(false, a3, false, bf, (short)0, acc3, false, false);
        }
    }

    // LSTM epilogue: lane n = column n; rows r (+8 for hi half-wave)
    const int moff = (lane >= 16) ? 8 : 0;
#pragma unroll
    for (int r = 0; r < 8; ++r) {
        const int ch = mtile * 16 + moff + r;
        if (ch < R) {
            const float iv = hsig (acc0[r] + bias[ch]);
            const float fv = hsig (acc1[r] + bias[R + ch]);
            const float gv = tanhf(acc2[r] + bias[2 * R + ch]);
            const float ov = hsig (acc3[r] + bias[3 * R + ch]);
            const size_t idx = ((size_t)b * R + ch) * HW + hw;
            const float cn = fv * cell[idx] + iv * gv;
            cell[idx]   = cn;
            repOut[idx] = (_Float16)(ov * tanhf(cn));
        }
    }
}

// ---------------------------------------------------------------------------
// Fused Ahat + Error: Ahat = relu(conv3x3(Rep)+b); layer0 clamps & writes the
// f32 prediction; Err = [relu(Ahat-A), relu(A-Ahat)] in f16.
// ---------------------------------------------------------------------------
__global__ __launch_bounds__(32)
void ahat_err_wmma(const _Float16* __restrict__ rep,
                   const _Float16* __restrict__ packedW,   // [MT][nCh][32][16]
                   const float* __restrict__ bias,         // [S]
                   const float* __restrict__ Af,           // layer0 target (x slice) or null
                   const _Float16* __restrict__ Ah,        // higher-layer target or null
                   long a_bstride,
                   _Float16* __restrict__ errOut,          // [B, 2S, H, W]
                   float* __restrict__ pred, long pred_bstride,
                   int Sch, int Rin, int gp0, int cpt, int H, int W, int clampPred)
{
    const int lane  = threadIdx.x;
    const int HW    = H * W;
    const int ptile = blockIdx.x;
    const int mtile = blockIdx.y;
    const int nChunks = 9 * cpt;

    const int n       = lane & 15;
    const int halfIdx = (lane >= 16) ? 1 : 0;
    const int p0 = ptile * 16;
    const int b  = p0 / HW;
    const int hw = (p0 - b * HW) + n;
    const int h  = hw / W, w = hw - h * W;

    const size_t strF = (size_t)HW;
    const _Float16* base0 = rep + (size_t)b * Rin * strF;

    v8f acc = {};
    for (int tap = 0; tap < 9; ++tap) {
        const int ky = (tap >= 6) ? 2 : (tap >= 3) ? 1 : 0;
        const int kx = tap - ky * 3;
        const int hh = h + ky - 1, ww = w + kx - 1;
        const bool inb = ((unsigned)hh < (unsigned)H) && ((unsigned)ww < (unsigned)W);
        const size_t pixF = (size_t)hh * W + ww;

#pragma unroll 2
        for (int sp = 0; sp < cpt; ++sp) {
            const int ck = tap * cpt + sp;
            const int sg = 2 * sp + halfIdx;
            v16h bf = {};
            if (inb && sg < gp0) {
                const _Float16* sptr = base0 + (size_t)(sg * 16) * strF + pixF;
                int nvalid = Rin - sg * 16;
                if (nvalid > 16) nvalid = 16;
#pragma unroll
                for (int e = 0; e < 16; ++e)
                    if (e < nvalid) bf[e] = sptr[(size_t)e * strF];
            }
            const v16h af = load_packed_a(packedW, mtile, nChunks, ck, lane);
            acc = __builtin_amdgcn_wmma_f32_16x16x32_f16(false, af, false, bf, (short)0, acc, false, false);
        }
    }

    const int moff = (lane >= 16) ? 8 : 0;
#pragma unroll
    for (int r = 0; r < 8; ++r) {
        const int ch = mtile * 16 + moff + r;
        if (ch < Sch) {
            float ahat = fmaxf(acc[r] + bias[ch], 0.f);
            if (clampPred) ahat = fminf(ahat, 1.f);
            const size_t aidx = (size_t)b * a_bstride + (size_t)ch * HW + hw;
            if (pred) pred[(size_t)b * pred_bstride + (size_t)ch * HW + hw] = ahat;
            const float a = Af ? Af[aidx] : (float)Ah[aidx];
            errOut[((size_t)b * (2 * Sch) + ch)       * HW + hw] = (_Float16)fmaxf(ahat - a, 0.f);
            errOut[((size_t)b * (2 * Sch) + Sch + ch) * HW + hw] = (_Float16)fmaxf(a - ahat, 0.f);
        }
    }
}

// ---------------------------------------------------------------------------
// A-path conv: out = relu(conv3x3(Err)+b) in f16 (pooled by maxpool2_f16)
// ---------------------------------------------------------------------------
__global__ __launch_bounds__(32)
void aconv_relu_wmma(const _Float16* __restrict__ src,     // [B, Cin, H, W]
                     const _Float16* __restrict__ packedW, // [MT][nCh][32][16]
                     const float* __restrict__ bias,       // [So]
                     _Float16* __restrict__ out,           // [B, So, H, W]
                     int So, int Cin, int gp0, int cpt, int H, int W)
{
    const int lane  = threadIdx.x;
    const int HW    = H * W;
    const int ptile = blockIdx.x;
    const int mtile = blockIdx.y;
    const int nChunks = 9 * cpt;

    const int n       = lane & 15;
    const int halfIdx = (lane >= 16) ? 1 : 0;
    const int p0 = ptile * 16;
    const int b  = p0 / HW;
    const int hw = (p0 - b * HW) + n;
    const int h  = hw / W, w = hw - h * W;

    const size_t strF = (size_t)HW;
    const _Float16* base0 = src + (size_t)b * Cin * strF;

    v8f acc = {};
    for (int tap = 0; tap < 9; ++tap) {
        const int ky = (tap >= 6) ? 2 : (tap >= 3) ? 1 : 0;
        const int kx = tap - ky * 3;
        const int hh = h + ky - 1, ww = w + kx - 1;
        const bool inb = ((unsigned)hh < (unsigned)H) && ((unsigned)ww < (unsigned)W);
        const size_t pixF = (size_t)hh * W + ww;

#pragma unroll 2
        for (int sp = 0; sp < cpt; ++sp) {
            const int ck = tap * cpt + sp;
            const int sg = 2 * sp + halfIdx;
            v16h bf = {};
            if (inb && sg < gp0) {
                const _Float16* sptr = base0 + (size_t)(sg * 16) * strF + pixF;
                int nvalid = Cin - sg * 16;
                if (nvalid > 16) nvalid = 16;
#pragma unroll
                for (int e = 0; e < 16; ++e)
                    if (e < nvalid) bf[e] = sptr[(size_t)e * strF];
            }
            const v16h af = load_packed_a(packedW, mtile, nChunks, ck, lane);
            acc = __builtin_amdgcn_wmma_f32_16x16x32_f16(false, af, false, bf, (short)0, acc, false, false);
        }
    }

    const int moff = (lane >= 16) ? 8 : 0;
#pragma unroll
    for (int r = 0; r < 8; ++r) {
        const int ch = mtile * 16 + moff + r;
        if (ch < So)
            out[((size_t)b * So + ch) * HW + hw] = (_Float16)fmaxf(acc[r] + bias[ch], 0.f);
    }
}

__global__ void maxpool2_f16(const _Float16* __restrict__ in, _Float16* __restrict__ out,
                             int H, int W, int total)
{
    const int i = blockIdx.x * blockDim.x + threadIdx.x;
    if (i >= total) return;
    const int W2 = W >> 1, H2 = H >> 1;
    const int w2 = i % W2;
    const int t  = i / W2;
    const int h2 = t % H2;
    const int bc = t / H2;
    const _Float16* s = in + (size_t)bc * H * W + (size_t)(2 * h2) * W + 2 * w2;
    const float m = fmaxf(fmaxf((float)s[0], (float)s[1]),
                          fmaxf((float)s[W], (float)s[W + 1]));
    out[i] = (_Float16)m;
}

__global__ void zero_u32(unsigned* __restrict__ p, size_t n)
{
    size_t i = (size_t)blockIdx.x * blockDim.x + threadIdx.x;
    const size_t stride = (size_t)gridDim.x * blockDim.x;
    for (; i < n; i += stride) p[i] = 0u;
}

// ---------------------------------------------------------------------------
// Host orchestration
// ---------------------------------------------------------------------------
extern "C" void kernel_launch(void* const* d_in, const int* in_sizes, int n_in,
                              void* d_out, int out_size, void* d_ws, size_t ws_size,
                              hipStream_t stream)
{
    (void)in_sizes; (void)n_in; (void)out_size; (void)ws_size;

    constexpr int L  = 4;
    constexpr int Bn = 8;
    constexpr int T  = 10;
    const int Hs[L] = {128, 64, 32, 16};
    const int Wd[L] = {160, 80, 40, 20};
    const int SC[L] = {3, 48, 96, 192};      // STACK == RSTACK
    int HW[L];
    for (int l = 0; l < L; ++l) HW[l] = Hs[l] * Wd[l];
    auto g16  = [](int n) { return (n + 15) / 16; };
    auto pad2 = [](int n) { return (n + 1) & ~1; };

    // --- inputs: x, then per layer {cell_W, cell_b, ahat_W, ahat_b, [a_W, a_b]}
    const float* x = (const float*)d_in[0];
    const float *cellW[L], *cellB[L], *ahatW[L], *ahatB[L], *aW[L], *aB[L];
    for (int l = 0; l < L; ++l) {
        const int base = 1 + l * 6;
        cellW[l] = (const float*)d_in[base + 0];
        cellB[l] = (const float*)d_in[base + 1];
        ahatW[l] = (const float*)d_in[base + 2];
        ahatB[l] = (const float*)d_in[base + 3];
        if (l < L - 1) { aW[l] = (const float*)d_in[base + 4]; aB[l] = (const float*)d_in[base + 5]; }
        else           { aW[l] = nullptr; aB[l] = nullptr; }
    }
    float* out = (float*)d_out;

    // --- workspace carve (bytes, 256B aligned chunks)
    char* basep = (char*)d_ws;
    size_t off = 0;
    auto carve = [&](size_t bytes) -> void* {
        off = (off + 255) & ~(size_t)255;
        void* p = basep + off;
        off += bytes;
        return p;
    };

    float*    cellF[L];
    _Float16 *repA[L], *repB[L], *errH[L], *abufH[L];
    for (int l = 0; l < L; ++l) cellF[l] = (float*)   carve(sizeof(float)    * (size_t)Bn * SC[l] * HW[l]);
    for (int l = 0; l < L; ++l) repA[l]  = (_Float16*)carve(sizeof(_Float16) * (size_t)Bn * SC[l] * HW[l]);
    for (int l = 0; l < L; ++l) errH[l]  = (_Float16*)carve(sizeof(_Float16) * (size_t)Bn * 2 * SC[l] * HW[l]);
    const size_t zeroBytes = (off + 3) & ~(size_t)3;      // Cell + Rep(cur) + Err start at 0
    for (int l = 0; l < L; ++l) repB[l]  = (_Float16*)carve(sizeof(_Float16) * (size_t)Bn * SC[l] * HW[l]);
    abufH[0] = nullptr;
    for (int l = 1; l < L; ++l) abufH[l] = (_Float16*)carve(sizeof(_Float16) * (size_t)Bn * SC[l] * HW[l]);
    _Float16* tmpH = (_Float16*)carve(sizeof(_Float16) * (size_t)Bn * SC[1] * HW[0]);  // largest A-path conv

    // --- packed weights (f16, fragment order, tap-major padded K')
    int cMT[L], cCPT[L], cGP[L][3], cGPT[L];
    int hMT[L], hCPT[L], hGP[L];
    int aMT[L], aCPT[L], aGP[L];
    _Float16 *cellPW[L], *ahatPW[L], *aPW[L];
    for (int l = 0; l < L; ++l) {
        const int n0 = SC[l], n1 = 2 * SC[l], n2 = (l < L - 1) ? SC[l + 1] : 0;
        cGP[l][0] = g16(n0); cGP[l][1] = g16(n1); cGP[l][2] = n2 ? g16(n2) : 0;
        cGPT[l] = cGP[l][0] + cGP[l][1] + cGP[l][2];
        cCPT[l] = pad2(cGPT[l]) / 2;
        cMT[l]  = g16(SC[l]);
        cellPW[l] = (_Float16*)carve(sizeof(_Float16) * (size_t)4 * cMT[l] * 9 * cCPT[l] * 512);

        hGP[l]  = g16(SC[l]);
        hCPT[l] = pad2(hGP[l]) / 2;
        hMT[l]  = g16(SC[l]);
        ahatPW[l] = (_Float16*)carve(sizeof(_Float16) * (size_t)hMT[l] * 9 * hCPT[l] * 512);

        if (l < L - 1) {
            aGP[l]  = g16(2 * SC[l]);
            aCPT[l] = pad2(aGP[l]) / 2;
            aMT[l]  = g16(SC[l + 1]);
            aPW[l]  = (_Float16*)carve(sizeof(_Float16) * (size_t)aMT[l] * 9 * aCPT[l] * 512);
        } else { aGP[l] = aCPT[l] = aMT[l] = 0; aPW[l] = nullptr; }
    }

    // --- init: zero recurrent state, pack all weights (deterministic each call)
    zero_u32<<<2048, 256, 0, stream>>>((unsigned*)d_ws, zeroBytes / 4);
    for (int l = 0; l < L; ++l) {
        const int n0 = SC[l], n1 = 2 * SC[l], n2 = (l < L - 1) ? SC[l + 1] : 0;
        int nc = 9 * cCPT[l];
        int tot = 4 * cMT[l] * nc * 512;
        pack_w_kernel<<<(tot + 255) / 256, 256, 0, stream>>>(
            cellW[l], cellPW[l], 4, SC[l], cMT[l],
            n0, n1, n2, cGP[l][0], cGP[l][1], cGPT[l], 2 * cCPT[l], nc, tot);
        nc = 9 * hCPT[l];
        tot = hMT[l] * nc * 512;
        pack_w_kernel<<<(tot + 255) / 256, 256, 0, stream>>>(
            ahatW[l], ahatPW[l], 1, SC[l], hMT[l],
            SC[l], 0, 0, hGP[l], 0, hGP[l], 2 * hCPT[l], nc, tot);
        if (l < L - 1) {
            nc = 9 * aCPT[l];
            tot = aMT[l] * nc * 512;
            pack_w_kernel<<<(tot + 255) / 256, 256, 0, stream>>>(
                aW[l], aPW[l], 1, SC[l + 1], aMT[l],
                2 * SC[l], 0, 0, aGP[l], 0, aGP[l], 2 * aCPT[l], nc, tot);
        }
    }

    // --- recurrence
    _Float16** cur = repA;
    _Float16** nxt = repB;
    const long predB = (long)T * 3 * HW[0];

    for (int t = 0; t < T; ++t) {
        // top-down ConvLSTM sweep (layer l consumes updated Rep[l+1])
        for (int l = L - 1; l >= 0; --l) {
            const int R = SC[l], S2 = 2 * SC[l];
            const int Rt = (l < L - 1) ? SC[l + 1] : 0;
            dim3 grid(Bn * HW[l] / 16, cMT[l]);
            convlstm_wmma<<<grid, 32, 0, stream>>>(
                cur[l], errH[l], (l < L - 1) ? nxt[l + 1] : nullptr,
                cellPW[l], cellB[l], cellF[l], nxt[l],
                R, S2, Rt, cGP[l][0], cGP[l][1], cGPT[l], cCPT[l],
                Hs[l], Wd[l], cMT[l]);
        }
        // bottom-up: Ahat + Error, then A-path conv + maxpool
        for (int l = 0; l < L; ++l) {
            const float*    Af = nullptr;
            const _Float16* Ah = nullptr;
            long abstr;
            if (l == 0) { Af = x + (size_t)t * 3 * HW[0]; abstr = (long)T * 3 * HW[0]; }
            else        { Ah = abufH[l];                  abstr = (long)SC[l] * HW[l]; }
            float* pred = (l == 0) ? out + (size_t)t * 3 * HW[0] : nullptr;
            dim3 grid(Bn * HW[l] / 16, hMT[l]);
            ahat_err_wmma<<<grid, 32, 0, stream>>>(
                nxt[l], ahatPW[l], ahatB[l], Af, Ah, abstr,
                errH[l], pred, predB,
                SC[l], SC[l], hGP[l], hCPT[l], Hs[l], Wd[l], l == 0 ? 1 : 0);

            if (l < L - 1) {
                const int So = SC[l + 1];
                dim3 gridc(Bn * HW[l] / 16, aMT[l]);
                aconv_relu_wmma<<<gridc, 32, 0, stream>>>(
                    errH[l], aPW[l], aB[l], tmpH, So, 2 * SC[l], aGP[l], aCPT[l], Hs[l], Wd[l]);
                const int total = Bn * So * (Hs[l] / 2) * (Wd[l] / 2);
                maxpool2_f16<<<(total + 255) / 256, 256, 0, stream>>>(
                    tmpH, abufH[l + 1], Hs[l], Wd[l], total);
            }
        }
        _Float16** s = cur; cur = nxt; nxt = s;
    }
}
